// Block_7636451852627
// MI455X (gfx1250) — matrix operand order repaired and checked by reading, not compiled, not run on hardware
//
#include <hip/hip_runtime.h>
#include <hip/hip_bf16.h>
#include <math.h>

typedef __attribute__((ext_vector_type(16))) _Float16 v16h;
typedef __attribute__((ext_vector_type(8)))  _Float16 v8h;
typedef __attribute__((ext_vector_type(8)))  float    v8f;

static constexpr int BSZ = 4, T = 2048, C = 1024, H = 16, HS = 64, CF = 4096;
static constexpr int M_TOK = BSZ * T;  // 8192 token rows

union AFrag { v16h v; v8h h[2]; };

// ---------------------------------------------------------------------------
// LayerNorm: one block (256 threads) per row of C=1024 floats -> f16 output
// ---------------------------------------------------------------------------
__global__ void ln_kernel(const float* __restrict__ x, const float* __restrict__ g,
                          const float* __restrict__ beta, _Float16* __restrict__ out) {
  __shared__ float red[2][8];
  const int row = blockIdx.x;
  const float* xr = x + (size_t)row * C;
  float vals[4], s = 0.f, s2 = 0.f;
#pragma unroll
  for (int i = 0; i < 4; i++) {
    float v = xr[threadIdx.x + i * 256];
    vals[i] = v; s += v; s2 += v * v;
  }
#pragma unroll
  for (int off = 16; off >= 1; off >>= 1) {
    s  += __shfl_xor(s,  off, 32);
    s2 += __shfl_xor(s2, off, 32);
  }
  const int w = threadIdx.x >> 5, lane = threadIdx.x & 31;
  if (lane == 0) { red[0][w] = s; red[1][w] = s2; }
  __syncthreads();
  s = 0.f; s2 = 0.f;
#pragma unroll
  for (int i = 0; i < 8; i++) { s += red[0][i]; s2 += red[1][i]; }
  const float mu = s * (1.0f / C);
  const float var = s2 * (1.0f / C) - mu * mu;
  const float rs = rsqrtf(var + 1e-5f);
#pragma unroll
  for (int i = 0; i < 4; i++) {
    int c = threadIdx.x + i * 256;
    out[(size_t)row * C + c] = (_Float16)((vals[i] - mu) * rs * g[c] + beta[c]);
  }
}

// ---------------------------------------------------------------------------
// Weight converts: f32 -> f16, transposed to [N][K] so B-fragments load
// 16 contiguous K halves per lane.
// ---------------------------------------------------------------------------
__global__ void cvt_head_w(const float* __restrict__ in, _Float16* __restrict__ out) {
  int idx = blockIdx.x * 256 + threadIdx.x;          // over 1024*1024
  int n = idx >> 10, c = idx & 1023;
  int h = n >> 6, d = n & 63;
  out[idx] = (_Float16)in[((size_t)h * C + c) * HS + d];
}
__global__ void cvt_mat_t(const float* __restrict__ in, _Float16* __restrict__ out,
                          int K, int N) {
  int idx = blockIdx.x * 256 + threadIdx.x;          // over N*K
  int n = idx / K, k = idx - n * K;
  out[idx] = (_Float16)in[(size_t)k * N + n];
}

// ---------------------------------------------------------------------------
// WMMA GEMM: C[M,N] = A[M,K](f16) * Bt[N,K](f16).
// Block = 8 waves, block tile 256(M) x 64(N); each wave owns a 32x64 tile.
// EPI: 0 = q/k head store [B,H,T,HS] (x oscale)  1 = v store d-major [B,H,HS,T]
//      2 = + bias + residual -> f32              3 = + bias, exact GELU -> f16
// ---------------------------------------------------------------------------
template <int EPI>
__global__ __launch_bounds__(256, 1)
void gemm_wmma(const _Float16* __restrict__ A, const _Float16* __restrict__ Bt,
               int M, int N, int K,
               const float* __restrict__ bias,
               const float* __restrict__ residual,
               float* __restrict__ outF,
               _Float16* __restrict__ outH,
               float oscale) {
  const int lane = threadIdx.x & 31;
  const int w    = threadIdx.x >> 5;
  const int l16  = lane & 15;
  const int hi   = lane >> 4;
  const int mbase = blockIdx.y * 256 + w * 32;
  const int nbase = blockIdx.x * 64;

  v8f acc[2][4] = {};
  const _Float16* arow0 = A + (size_t)(mbase + l16) * K;
  const _Float16* arow1 = A + (size_t)(mbase + 16 + l16) * K;
  const _Float16* brow  = Bt + (size_t)(nbase + l16) * K;
  const int aoff0 = hi * 16, aoff1 = hi * 16 + 8, boff = hi * 16;

  for (int k0 = 0; k0 < K; k0 += 32) {
    AFrag a[2], b[4];
    a[0].h[0] = *(const v8h*)(arow0 + k0 + aoff0);
    a[0].h[1] = *(const v8h*)(arow0 + k0 + aoff1);
    a[1].h[0] = *(const v8h*)(arow1 + k0 + aoff0);
    a[1].h[1] = *(const v8h*)(arow1 + k0 + aoff1);
#pragma unroll
    for (int j = 0; j < 4; j++) {
      const _Float16* bp = brow + (size_t)j * 16 * K + k0 + boff;
      b[j].h[0] = *(const v8h*)(bp);
      b[j].h[1] = *(const v8h*)(bp + 8);
    }
    if (k0 + 32 < K) {  // gfx1250 global_prefetch_b8 on next k-chunk
      __builtin_prefetch(arow0 + k0 + 32 + aoff0, 0, 0);
      __builtin_prefetch(arow1 + k0 + 32 + aoff0, 0, 0);
      __builtin_prefetch(brow + k0 + 32 + boff, 0, 0);
    }
#pragma unroll
    for (int j = 0; j < 4; j++) {
#pragma unroll
      for (int i = 0; i < 2; i++) {
        acc[i][j] = __builtin_amdgcn_wmma_f32_16x16x32_f16(
            false, a[i].v, false, b[j].v, (short)0, acc[i][j], false, false);
      }
    }
  }

#pragma unroll
  for (int i = 0; i < 2; i++) {
#pragma unroll
    for (int j = 0; j < 4; j++) {
#pragma unroll
      for (int r = 0; r < 8; r++) {
        const int row = mbase + i * 16 + r + hi * 8;
        const int col = nbase + j * 16 + l16;
        float v = acc[i][j][r];
        if (EPI == 0) {
          int b_ = row >> 11, t = row & (T - 1);
          int h = col >> 6, d = col & 63;
          outH[(((size_t)(b_ * H + h)) * T + t) * HS + d] = (_Float16)(v * oscale);
        } else if (EPI == 1) {
          int b_ = row >> 11, t = row & (T - 1);
          int h = col >> 6, d = col & 63;
          outH[(((size_t)(b_ * H + h)) * HS + d) * T + t] = (_Float16)(v * oscale);
        } else if (EPI == 2) {
          outF[(size_t)row * N + col] = v + bias[col] + residual[(size_t)row * N + col];
        } else {
          float u = v + bias[col];
          outH[(size_t)row * N + col] = (_Float16)(0.5f * u * (1.0f + erff(u * 0.70710678f)));
        }
      }
    }
  }
}

// ---------------------------------------------------------------------------
// Flash attention key-tile processor. MASKED only for the diagonal tail tile.
// Softmax scale is pre-folded into Q (exact: 0.125 = 2^-3).
// ---------------------------------------------------------------------------
template <bool MASKED>
__device__ __forceinline__ void fa_tile(int s0, int qbase, int l16, int hi, int w,
                                        const _Float16* __restrict__ kp,
                                        const _Float16* __restrict__ vp,
                                        const AFrag (&qa)[2], v8f (&o)[4],
                                        float (&m)[8], float (&l)[8],
                                        _Float16 (*plds)[16][64]) {
  // S = Q K^T over 64 keys
  v8f sf[4] = {};
#pragma unroll
  for (int j = 0; j < 4; j++) {
#pragma unroll
    for (int kk = 0; kk < 2; kk++) {
      AFrag bf;
      const _Float16* bp = kp + (size_t)(s0 + j * 16 + l16) * HS + kk * 32 + hi * 16;
      bf.h[0] = *(const v8h*)bp;
      bf.h[1] = *(const v8h*)(bp + 8);
      sf[j] = __builtin_amdgcn_wmma_f32_16x16x32_f16(false, qa[kk].v, false, bf.v,
                                                     (short)0, sf[j], false, false);
    }
  }
  // per-row: (mask) -> row max -> online softmax -> P to LDS
#pragma unroll
  for (int r = 0; r < 8; r++) {
    const int t = qbase + r + hi * 8;
    float mx = -3.0e38f;
#pragma unroll
    for (int j = 0; j < 4; j++) {
      float v = sf[j][r];
      if (MASKED) {
        const int s = s0 + j * 16 + l16;
        v = (s <= t) ? v : -3.0e38f;
        sf[j][r] = v;
      }
      mx = fmaxf(mx, v);
    }
#pragma unroll
    for (int off = 1; off < 16; off <<= 1) mx = fmaxf(mx, __shfl_xor(mx, off, 32));
    const float mn = fmaxf(m[r], mx);
    const float alpha = expf(m[r] - mn);
    m[r] = mn;
    float rs = 0.f;
#pragma unroll
    for (int j = 0; j < 4; j++) {
      float p = expf(sf[j][r] - mn);
      rs += p;
      plds[w][r + hi * 8][j * 16 + l16] = (_Float16)p;
    }
#pragma unroll
    for (int off = 1; off < 16; off <<= 1) rs += __shfl_xor(rs, off, 32);
    l[r] = l[r] * alpha + rs;
#pragma unroll
    for (int j = 0; j < 4; j++) o[j][r] *= alpha;
  }
  __asm volatile("s_wait_dscnt 0" ::: "memory");  // P stores visible to A-frag loads
  // O += P * V  (A = P from LDS, B = V^T [d][T] contiguous along s)
#pragma unroll
  for (int kk = 0; kk < 2; kk++) {
    AFrag pa;
    const _Float16* pr = &plds[w][l16][0];
    pa.h[0] = *(const v8h*)(pr + kk * 32 + hi * 16);
    pa.h[1] = *(const v8h*)(pr + kk * 32 + hi * 16 + 8);
#pragma unroll
    for (int j = 0; j < 4; j++) {
      AFrag bf;
      const _Float16* bp = vp + (size_t)(j * 16 + l16) * T + s0 + kk * 32 + hi * 16;
      bf.h[0] = *(const v8h*)bp;
      bf.h[1] = *(const v8h*)(bp + 8);
      o[j] = __builtin_amdgcn_wmma_f32_16x16x32_f16(false, pa.v, false, bf.v,
                                                    (short)0, o[j], false, false);
    }
  }
  __asm volatile("s_wait_dscnt 0" ::: "memory");  // LDS reads done before next overwrite
}

// ---------------------------------------------------------------------------
// Flash attention, causal. grid = (T/128, B*H), block = 256 (8 waves).
// Each wave owns 16 query rows; key tiles of 64. Q(pre-scaled),K: [BH][T][HS],
// V pre-transposed: [BH][HS][T] f16. Output -> attn [B*T][C] f16.
// ---------------------------------------------------------------------------
__global__ __launch_bounds__(256, 1)
void flash_kernel(const _Float16* __restrict__ Q,
                  const _Float16* __restrict__ Kh,
                  const _Float16* __restrict__ Vt,
                  _Float16* __restrict__ attnOut) {
  __shared__ __align__(32) _Float16 plds[8][16][64];  // per-wave P tile, 16 KB
  const int lane = threadIdx.x & 31;
  const int w    = threadIdx.x >> 5;
  const int l16  = lane & 15;
  const int hi   = lane >> 4;
  const int bh = blockIdx.y;
  const int b = bh >> 4, h = bh & 15;
  const int qbase = blockIdx.x * 128 + w * 16;

  const _Float16* qp = Q  + ((size_t)bh * T + qbase) * HS;
  const _Float16* kp = Kh + (size_t)bh * T * HS;
  const _Float16* vp = Vt + (size_t)bh * HS * T;

  // Q fragments (K-dim = HS = 64 -> 2 A-frags), resident for the whole pass
  AFrag qa[2];
  {
    const _Float16* qr = qp + (size_t)l16 * HS;
#pragma unroll
    for (int kk = 0; kk < 2; kk++) {
      qa[kk].h[0] = *(const v8h*)(qr + kk * 32 + hi * 16);
      qa[kk].h[1] = *(const v8h*)(qr + kk * 32 + hi * 16 + 8);
    }
  }

  v8f o[4] = {};
  float m[8], l[8];
#pragma unroll
  for (int r = 0; r < 8; r++) { m[r] = -3.0e38f; l[r] = 0.f; }
  const int smax = qbase + 15;

  int s0 = 0;
  for (; s0 + 63 <= qbase; s0 += 64)   // full tiles: no causal masking needed
    fa_tile<false>(s0, qbase, l16, hi, w, kp, vp, qa, o, m, l, plds);
  for (; s0 <= smax; s0 += 64)         // diagonal tail tile(s): masked
    fa_tile<true>(s0, qbase, l16, hi, w, kp, vp, qa, o, m, l, plds);

  // normalize and store heads concatenated: attn[b*T+t][h*64+d]
#pragma unroll
  for (int j = 0; j < 4; j++) {
#pragma unroll
    for (int r = 0; r < 8; r++) {
      const int t = qbase + r + hi * 8;
      const int d = j * 16 + l16;
      attnOut[((size_t)(b * T + t)) * C + h * 64 + d] = (_Float16)(o[j][r] / l[r]);
    }
  }
}

// ---------------------------------------------------------------------------
// Host orchestration
// ---------------------------------------------------------------------------
extern "C" void kernel_launch(void* const* d_in, const int* in_sizes, int n_in,
                              void* d_out, int out_size, void* d_ws, size_t ws_size,
                              hipStream_t stream) {
  const float* x     = (const float*)d_in[0];
  const float* Wq    = (const float*)d_in[1];
  const float* Wk    = (const float*)d_in[2];
  const float* Wv    = (const float*)d_in[3];
  const float* Wp    = (const float*)d_in[4];
  const float* bp    = (const float*)d_in[5];
  const float* W1    = (const float*)d_in[6];
  const float* b1    = (const float*)d_in[7];
  const float* W2    = (const float*)d_in[8];
  const float* b2    = (const float*)d_in[9];
  const float* g1    = (const float*)d_in[10];
  const float* beta1 = (const float*)d_in[11];
  const float* g2    = (const float*)d_in[12];
  const float* beta2 = (const float*)d_in[13];

  char* ws = (char*)d_ws;
  const size_t MB = 1ull << 20;
  _Float16* h16  = (_Float16*)(ws + 0 * MB);     // 16 MB  [8192][1024]
  _Float16* wqT  = (_Float16*)(ws + 16 * MB);    //  2 MB
  _Float16* wkT  = (_Float16*)(ws + 18 * MB);    //  2 MB
  _Float16* wvT  = (_Float16*)(ws + 20 * MB);    //  2 MB
  _Float16* wpT  = (_Float16*)(ws + 22 * MB);    //  2 MB
  _Float16* w1T  = (_Float16*)(ws + 24 * MB);    //  8 MB  [4096][1024]
  _Float16* w2T  = (_Float16*)(ws + 32 * MB);    //  8 MB  [1024][4096]
  _Float16* qb   = (_Float16*)(ws + 40 * MB);    // 16 MB  [BH][T][64]
  _Float16* kb   = (_Float16*)(ws + 56 * MB);    // 16 MB  [BH][T][64]
  _Float16* vtb  = (_Float16*)(ws + 72 * MB);    // 16 MB  [BH][64][T]
  _Float16* attn = (_Float16*)(ws + 88 * MB);    // 16 MB  [8192][1024]
  float*    x2   = (float*)   (ws + 104 * MB);   // 32 MB  [8192][1024]
  _Float16* h2   = (_Float16*)(ws + 136 * MB);   // 16 MB
  _Float16* ff1  = (_Float16*)(ws + 152 * MB);   // 64 MB  [8192][4096]

  // 1. LN1
  ln_kernel<<<M_TOK, 256, 0, stream>>>(x, g1, beta1, h16);

  // 2. weight conversion / transposition to [N][K] f16
  cvt_head_w<<<(C * C) / 256, 256, 0, stream>>>(Wq, wqT);
  cvt_head_w<<<(C * C) / 256, 256, 0, stream>>>(Wk, wkT);
  cvt_head_w<<<(C * C) / 256, 256, 0, stream>>>(Wv, wvT);
  cvt_mat_t<<<(C * C) / 256, 256, 0, stream>>>(Wp, wpT, C, C);
  cvt_mat_t<<<(C * CF) / 256, 256, 0, stream>>>(W1, w1T, C, CF);
  cvt_mat_t<<<(CF * C) / 256, 256, 0, stream>>>(W2, w2T, CF, C);

  // 3. QKV projections (WMMA GEMMs). Q gets the softmax scale folded in (2^-3).
  dim3 gQKV(C / 64, M_TOK / 256);
  gemm_wmma<0><<<gQKV, 256, 0, stream>>>(h16, wqT, M_TOK, C, C, nullptr, nullptr, nullptr, qb, 0.125f);
  gemm_wmma<0><<<gQKV, 256, 0, stream>>>(h16, wkT, M_TOK, C, C, nullptr, nullptr, nullptr, kb, 1.0f);
  gemm_wmma<1><<<gQKV, 256, 0, stream>>>(h16, wvT, M_TOK, C, C, nullptr, nullptr, nullptr, vtb, 1.0f);

  // 4. causal flash attention
  dim3 gFA(T / 128, BSZ * H);
  flash_kernel<<<gFA, 256, 0, stream>>>(qb, kb, vtb, attn);

  // 5. output projection + bias + residual -> x2 (f32)
  gemm_wmma<2><<<gQKV, 256, 0, stream>>>(attn, wpT, M_TOK, C, C, bp, x, x2, nullptr, 1.0f);

  // 6. LN2
  ln_kernel<<<M_TOK, 256, 0, stream>>>(x2, g2, beta2, h2);

  // 7. FFN up-projection + bias + exact GELU -> ff1 (f16)
  dim3 gF1(CF / 64, M_TOK / 256);
  gemm_wmma<3><<<gF1, 256, 0, stream>>>(h2, w1T, M_TOK, CF, C, b1, nullptr, nullptr, ff1, 1.0f);

  // 8. FFN down-projection + bias + residual -> d_out (f32)
  gemm_wmma<2><<<gQKV, 256, 0, stream>>>(ff1, w2T, M_TOK, C, CF, b2, x2, (float*)d_out, nullptr, 1.0f);
}
